// DotProductAttention_81037442941648
// MI455X (gfx1250) — matrix operand order repaired and checked by reading, not compiled, not run on hardware
//
#include <hip/hip_runtime.h>

typedef __attribute__((ext_vector_type(16))) _Float16 v16h;
typedef __attribute__((ext_vector_type(8)))  _Float16 v8h;
typedef __attribute__((ext_vector_type(4)))  _Float16 v4h;
typedef __attribute__((ext_vector_type(2)))  _Float16 v2h;
typedef __attribute__((ext_vector_type(8)))  float    v8f;

#define S_LEN   2048
#define DKK     64
#define DVV     64
#define QTILE   128         // q rows per workgroup (8 waves x 16)
#define KVT     64          // keys per inner iteration (2 WMMA contract chunks)
#define KSTR    72          // halves: 144B padded stride (K tile rows)
#define VSTR    72          // halves: 144B padded stride (V^T rows)
#define PSTR    72          // halves: 144B padded stride (P rows)
#define SSTR    68          // floats: 272B padded stride (S rows)
#define NBH     32          // B*H

__device__ __forceinline__ v8h cvt8s(const float* p, float s) {
    float4 f0 = *reinterpret_cast<const float4*>(p);
    float4 f1 = *reinterpret_cast<const float4*>(p + 4);
    v8h h;
    h[0]=(_Float16)(f0.x*s); h[1]=(_Float16)(f0.y*s); h[2]=(_Float16)(f0.z*s); h[3]=(_Float16)(f0.w*s);
    h[4]=(_Float16)(f1.x*s); h[5]=(_Float16)(f1.y*s); h[6]=(_Float16)(f1.z*s); h[7]=(_Float16)(f1.w*s);
    return h;
}
__device__ __forceinline__ v16h cat16(v8h lo, v8h hi) {
    return __builtin_shufflevector(lo, hi, 0,1,2,3,4,5,6,7,8,9,10,11,12,13,14,15);
}

// CDNA5 async global->LDS copy, GVS mode: mem = SADDR + VADDR(u32); LDS addr per lane.
__device__ __forceinline__ void async_b128(unsigned lds_off, unsigned gofs, const _Float16* base) {
    asm volatile("global_load_async_to_lds_b128 %0, %1, %2 offset:0"
                 :: "v"(lds_off), "v"(gofs), "s"(base) : "memory");
}
__device__ __forceinline__ void wait_async0() {
    asm volatile("s_wait_asynccnt 0x0" ::: "memory");
}

// =========================== pre-pass: K/V -> f16 ===========================
// K16  : [bh][s][dk]  row-major f16
// V16T : [bh][dv][s]  transposed f16
__global__ __launch_bounds__(256)
void cvt_kv_f16(const float* __restrict__ K, const float* __restrict__ V,
                _Float16* __restrict__ K16, _Float16* __restrict__ V16T)
{
    size_t tid = (size_t)blockIdx.x * 256 + threadIdx.x;   // one float4 per thread
    // K: straight conversion
    float4 kf = reinterpret_cast<const float4*>(K)[tid];
    v4h kh;
    kh[0]=(_Float16)kf.x; kh[1]=(_Float16)kf.y; kh[2]=(_Float16)kf.z; kh[3]=(_Float16)kf.w;
    reinterpret_cast<v4h*>(K16)[tid] = kh;
    // V: convert + transpose
    float4 vf = reinterpret_cast<const float4*>(V)[tid];
    int    d0 = (int)(tid & 15) * 4;
    size_t s  = (tid >> 4) & (S_LEN - 1);
    size_t bh = tid >> 15;                                  // 16 float4 * 2048 rows
    _Float16* vt = V16T + bh * (size_t)S_LEN * DVV;
    vt[(size_t)(d0+0) * S_LEN + s] = (_Float16)vf.x;
    vt[(size_t)(d0+1) * S_LEN + s] = (_Float16)vf.y;
    vt[(size_t)(d0+2) * S_LEN + s] = (_Float16)vf.z;
    vt[(size_t)(d0+3) * S_LEN + s] = (_Float16)vf.w;
}

// ===================== main kernel (async staging path) =====================
__global__ __launch_bounds__(256)
void fa_causal_f16wmma_async(const float* __restrict__ Q,
                             const _Float16* __restrict__ K16,
                             const _Float16* __restrict__ V16T,
                             float* __restrict__ O)
{
    const int lane  = threadIdx.x & 31;
    const int wv    = threadIdx.x >> 5;
    const int bh    = blockIdx.x >> 4;
    const int qt    = blockIdx.x & 15;
    const int qbase = qt * QTILE;
    const int qrow0 = qbase + wv * 16;

    const float*    Qh  = Q    + (size_t)bh * S_LEN * DKK;
    const _Float16* Kh  = K16  + (size_t)bh * S_LEN * DKK;
    const _Float16* Vth = V16T + (size_t)bh * S_LEN * DVV;
    float*          Oh  = O    + (size_t)bh * S_LEN * DVV;

    const int m16 = lane & 15;
    const int g   = lane >> 4;

    __shared__ _Float16 Kt[2][KVT * KSTR];
    __shared__ _Float16 Vt[2][DVV * VSTR];
    __shared__ _Float16 Pst[8][16 * PSTR];
    __shared__ float    Ssc[8][16 * SSTR];
    __shared__ float    Cst[8][16];

    // async staging granules: 64 rows x 8 b128-granules per tensor, 2 per thread
    const int gr0 = threadIdx.x;            // granule ids: gr0, gr0+256
    // per-granule coords (row, 8-half column chunk)
    const int kr0 = gr0 >> 3,            kc0 = (gr0 & 7) * 8;
    const int kr1 = (gr0 + 256) >> 3,    kc1 = kc0;   // same low bits
    const unsigned ldsK0[2] = {
        (unsigned)(size_t)&Kt[0][kr0 * KSTR + kc0], (unsigned)(size_t)&Kt[0][kr1 * KSTR + kc1] };
    const unsigned ldsK1[2] = {
        (unsigned)(size_t)&Kt[1][kr0 * KSTR + kc0], (unsigned)(size_t)&Kt[1][kr1 * KSTR + kc1] };
    const unsigned ldsV0[2] = {
        (unsigned)(size_t)&Vt[0][kr0 * VSTR + kc0], (unsigned)(size_t)&Vt[0][kr1 * VSTR + kc1] };
    const unsigned ldsV1[2] = {
        (unsigned)(size_t)&Vt[1][kr0 * VSTR + kc0], (unsigned)(size_t)&Vt[1][kr1 * VSTR + kc1] };

    // ---- Q tile in WMMA A layout, pre-scaled by (1/sqrt(dk))*log2(e) ----
    const float qs = 0.18033688011112042f;
    v16h aQ[2];
    {
        const float* qr = Qh + (size_t)(qrow0 + m16) * DKK;
#pragma unroll
        for (int c = 0; c < 2; ++c)
            aQ[c] = cat16(cvt8s(qr + c*32 + 16*g, qs), cvt8s(qr + c*32 + 16*g + 8, qs));
    }

    float mrow = -__builtin_inff();
    float lrow = 0.f;
    v8f oacc[4];
#pragma unroll
    for (int t = 0; t < 4; ++t) { v8f z = {}; oacc[t] = z; }

    const int grow = qrow0 + m16;
    const int nblk = 2*qt + 2;

    // ---- prologue: async-stage tile 0 into buffer 0 ----
    async_b128(ldsK0[0], (unsigned)((kr0 * DKK  + kc0) * 2), Kh);
    async_b128(ldsK0[1], (unsigned)((kr1 * DKK  + kc1) * 2), Kh);
    async_b128(ldsV0[0], (unsigned)((kr0 * S_LEN + kc0) * 2), Vth);
    async_b128(ldsV0[1], (unsigned)((kr1 * S_LEN + kc1) * 2), Vth);
    wait_async0();
    __syncthreads();

    for (int kb = 0; kb < nblk; ++kb) {
        const int cur  = kb & 1;
        const int key0 = kb * KVT;
        const bool have_next = (kb + 1 < nblk);

        // ---- issue next tile's async copies into the other buffer ----
        if (have_next) {
            const int keyn = key0 + KVT;
            const unsigned* lk = cur ? ldsK0 : ldsK1;
            const unsigned* lv = cur ? ldsV0 : ldsV1;
            async_b128(lk[0], (unsigned)(((keyn + kr0) * DKK + kc0) * 2), Kh);
            async_b128(lk[1], (unsigned)(((keyn + kr1) * DKK + kc1) * 2), Kh);
            async_b128(lv[0], (unsigned)((kr0 * S_LEN + keyn + kc0) * 2), Vth);
            async_b128(lv[1], (unsigned)((kr1 * S_LEN + keyn + kc1) * 2), Vth);
        }

        // ---- S' = (Q*qs) @ K^T ----
        v8f st[4];
#pragma unroll
        for (int nt = 0; nt < 4; ++nt) {
            const _Float16* krow = &Kt[cur][(nt*16 + m16) * KSTR];
            v16h b0 = cat16(*reinterpret_cast<const v8h*>(krow + 16*g),
                            *reinterpret_cast<const v8h*>(krow + 16*g + 8));
            v16h b1 = cat16(*reinterpret_cast<const v8h*>(krow + 32 + 16*g),
                            *reinterpret_cast<const v8h*>(krow + 32 + 16*g + 8));
            v8f cacc = {};
            cacc = __builtin_amdgcn_wmma_f32_16x16x32_f16(false, aQ[0], false, b0,
                                                          (short)0, cacc, false, false);
            cacc = __builtin_amdgcn_wmma_f32_16x16x32_f16(false, aQ[1], false, b1,
                                                          (short)0, cacc, false, false);
            st[nt] = cacc;
        }

        // ---- transpose S through LDS ----
        float* Sw = Ssc[wv];
#pragma unroll
        for (int nt = 0; nt < 4; ++nt)
#pragma unroll
            for (int r = 0; r < 8; ++r)
                Sw[(8*g + r) * SSTR + nt*16 + m16] = st[nt][r];
        __builtin_amdgcn_wave_barrier();

        const float* srow = &Sw[m16 * SSTR + 32*g];
        float v[32];
#pragma unroll
        for (int q4 = 0; q4 < 8; ++q4) {
            float4 x = *reinterpret_cast<const float4*>(srow + 4*q4);
            v[4*q4+0]=x.x; v[4*q4+1]=x.y; v[4*q4+2]=x.z; v[4*q4+3]=x.w;
        }

        if (key0 + (KVT - 1) > qrow0) {            // causal mask (uniform branch)
            const int c0 = key0 + 32*g;
#pragma unroll
            for (int j = 0; j < 32; ++j)
                if (c0 + j > grow) v[j] = -__builtin_inff();
        }

        float tm[16];
#pragma unroll
        for (int j = 0; j < 16; ++j) tm[j] = fmaxf(v[j], v[j+16]);
#pragma unroll
        for (int j = 0; j < 8; ++j)  tm[j] = fmaxf(tm[j], tm[j+8]);
#pragma unroll
        for (int j = 0; j < 4; ++j)  tm[j] = fmaxf(tm[j], tm[j+4]);
        float mx = fmaxf(fmaxf(tm[0], tm[1]), fmaxf(tm[2], tm[3]));
        mx = fmaxf(mx, __shfl_xor(mx, 16, 32));

        float mnew = fmaxf(mrow, mx);
#pragma unroll
        for (int j = 0; j < 32; ++j) v[j] = exp2f(v[j] - mnew);

        float ts[16];
#pragma unroll
        for (int j = 0; j < 16; ++j) ts[j] = v[j] + v[j+16];
#pragma unroll
        for (int j = 0; j < 8; ++j)  ts[j] = ts[j] + ts[j+8];
#pragma unroll
        for (int j = 0; j < 4; ++j)  ts[j] = ts[j] + ts[j+4];
        float ps = (ts[0] + ts[1]) + (ts[2] + ts[3]);
        ps += __shfl_xor(ps, 16, 32);

        float cr = exp2f(mrow - mnew);
        lrow = lrow * cr + ps;
        mrow = mnew;

        _Float16* Pw = Pst[wv];
#pragma unroll
        for (int q8 = 0; q8 < 4; ++q8) {
            v8h ph;
#pragma unroll
            for (int j = 0; j < 8; ++j) ph[j] = (_Float16)v[8*q8 + j];
            *reinterpret_cast<v8h*>(&Pw[m16 * PSTR + 32*g + 8*q8]) = ph;
        }

        if (g == 0) Cst[wv][m16] = cr;
        __builtin_amdgcn_wave_barrier();
        float4 ca = *reinterpret_cast<const float4*>(&Cst[wv][8*g]);
        float4 cb = *reinterpret_cast<const float4*>(&Cst[wv][8*g + 4]);
        float cf[8] = { ca.x,ca.y,ca.z,ca.w, cb.x,cb.y,cb.z,cb.w };
#pragma unroll
        for (int t = 0; t < 4; ++t)
#pragma unroll
            for (int r = 0; r < 8; ++r)
                oacc[t][r] *= cf[r];

        v16h aP[2];
#pragma unroll
        for (int c = 0; c < 2; ++c)
            aP[c] = cat16(*reinterpret_cast<const v8h*>(&Pw[m16 * PSTR + 32*c + 16*g]),
                          *reinterpret_cast<const v8h*>(&Pw[m16 * PSTR + 32*c + 16*g + 8]));

#pragma unroll
        for (int nt = 0; nt < 4; ++nt) {
            const _Float16* vrow = &Vt[cur][(nt*16 + m16) * VSTR];
            v16h b0 = cat16(*reinterpret_cast<const v8h*>(vrow + 16*g),
                            *reinterpret_cast<const v8h*>(vrow + 16*g + 8));
            v16h b1 = cat16(*reinterpret_cast<const v8h*>(vrow + 32 + 16*g),
                            *reinterpret_cast<const v8h*>(vrow + 32 + 16*g + 8));
            oacc[nt] = __builtin_amdgcn_wmma_f32_16x16x32_f16(false, aP[0], false, b0,
                                                              (short)0, oacc[nt], false, false);
            oacc[nt] = __builtin_amdgcn_wmma_f32_16x16x32_f16(false, aP[1], false, b1,
                                                              (short)0, oacc[nt], false, false);
        }

        wait_async0();     // next-tile async copies landed in LDS
        __syncthreads();   // visible to all waves; readers of buf[cur] done
    }

    if (g == 0) Cst[wv][m16] = 1.0f / lrow;
    __builtin_amdgcn_wave_barrier();
    float4 ia = *reinterpret_cast<const float4*>(&Cst[wv][8*g]);
    float4 ib = *reinterpret_cast<const float4*>(&Cst[wv][8*g + 4]);
    float inv[8] = { ia.x,ia.y,ia.z,ia.w, ib.x,ib.y,ib.z,ib.w };
#pragma unroll
    for (int nt = 0; nt < 4; ++nt)
#pragma unroll
        for (int r = 0; r < 8; ++r)
            Oh[(size_t)(qrow0 + 8*g + r) * DVV + nt*16 + m16] = oacc[nt][r] * inv[r];
}

// ============== fallback (fp32 inputs, register staging; round-5 path) ==============
__device__ __forceinline__ void stage_load(const float* __restrict__ Kh,
                                           const float* __restrict__ Vh,
                                           int key0, int u, int c4,
                                           float4 kr[4], float4 vr[4]) {
#pragma unroll
    for (int i = 0; i < 4; ++i)
        kr[i] = *reinterpret_cast<const float4*>(Kh + (size_t)(key0 + u + 16*i) * DKK + c4);
#pragma unroll
    for (int i = 0; i < 2; ++i) {
        int row = key0 + 2*u + 32*i;
        vr[2*i]   = *reinterpret_cast<const float4*>(Vh + (size_t)row * DVV + c4);
        vr[2*i+1] = *reinterpret_cast<const float4*>(Vh + (size_t)(row+1) * DVV + c4);
    }
}
__device__ __forceinline__ void stage_store(_Float16* Ktb, _Float16* Vtb,
                                            int u, int c4,
                                            const float4 kr[4], const float4 vr[4]) {
#pragma unroll
    for (int i = 0; i < 4; ++i) {
        v4h kh;
        kh[0]=(_Float16)kr[i].x; kh[1]=(_Float16)kr[i].y;
        kh[2]=(_Float16)kr[i].z; kh[3]=(_Float16)kr[i].w;
        *reinterpret_cast<v4h*>(&Ktb[(u + 16*i) * KSTR + c4]) = kh;
    }
#pragma unroll
    for (int i = 0; i < 2; ++i) {
        int row = 2*u + 32*i;
        const float4 a = vr[2*i], b = vr[2*i+1];
        float av[4] = { a.x, a.y, a.z, a.w };
        float bv[4] = { b.x, b.y, b.z, b.w };
#pragma unroll
        for (int j = 0; j < 4; ++j) {
            v2h p; p[0] = (_Float16)av[j]; p[1] = (_Float16)bv[j];
            *reinterpret_cast<v2h*>(&Vtb[(c4+j) * VSTR + row]) = p;
        }
    }
}

__global__ __launch_bounds__(256)
void fa_causal_f16wmma(const float* __restrict__ Q, const float* __restrict__ K,
                       const float* __restrict__ V, float* __restrict__ O)
{
    const int lane  = threadIdx.x & 31;
    const int wv    = threadIdx.x >> 5;
    const int bh    = blockIdx.x >> 4;
    const int qt    = blockIdx.x & 15;
    const int qbase = qt * QTILE;
    const int qrow0 = qbase + wv * 16;

    const float* Qh = Q + (size_t)bh * S_LEN * DKK;
    const float* Kh = K + (size_t)bh * S_LEN * DKK;
    const float* Vh = V + (size_t)bh * S_LEN * DVV;
    float*       Oh = O + (size_t)bh * S_LEN * DVV;

    const int m16 = lane & 15;
    const int g   = lane >> 4;

    __shared__ _Float16 Kt[2][KVT * KSTR];
    __shared__ _Float16 Vt[2][DVV * VSTR];
    __shared__ _Float16 Pst[8][16 * PSTR];
    __shared__ float    Ssc[8][16 * SSTR];
    __shared__ float    Cst[8][16];

    const int u  = threadIdx.x >> 4;
    const int c4 = (threadIdx.x & 15) * 4;

    const float qs = 0.18033688011112042f;
    v16h aQ[2];
    {
        const float* qr = Qh + (size_t)(qrow0 + m16) * DKK;
#pragma unroll
        for (int c = 0; c < 2; ++c)
            aQ[c] = cat16(cvt8s(qr + c*32 + 16*g, qs), cvt8s(qr + c*32 + 16*g + 8, qs));
    }

    float mrow = -__builtin_inff();
    float lrow = 0.f;
    v8f oacc[4];
#pragma unroll
    for (int t = 0; t < 4; ++t) { v8f z = {}; oacc[t] = z; }

    const int grow = qrow0 + m16;
    const int nblk = 2*qt + 2;

    {
        float4 kr[4], vr[4];
        stage_load(Kh, Vh, 0, u, c4, kr, vr);
        stage_store(Kt[0], Vt[0], u, c4, kr, vr);
    }
    __syncthreads();

    for (int kb = 0; kb < nblk; ++kb) {
        const int cur  = kb & 1;
        const int nxt  = cur ^ 1;
        const int key0 = kb * KVT;
        const bool have_next = (kb + 1 < nblk);

        float4 kr[4], vr[4];
        if (have_next) stage_load(Kh, Vh, key0 + KVT, u, c4, kr, vr);

        v8f st[4];
#pragma unroll
        for (int nt = 0; nt < 4; ++nt) {
            const _Float16* krow = &Kt[cur][(nt*16 + m16) * KSTR];
            v16h b0 = cat16(*reinterpret_cast<const v8h*>(krow + 16*g),
                            *reinterpret_cast<const v8h*>(krow + 16*g + 8));
            v16h b1 = cat16(*reinterpret_cast<const v8h*>(krow + 32 + 16*g),
                            *reinterpret_cast<const v8h*>(krow + 32 + 16*g + 8));
            v8f cacc = {};
            cacc = __builtin_amdgcn_wmma_f32_16x16x32_f16(false, aQ[0], false, b0,
                                                          (short)0, cacc, false, false);
            cacc = __builtin_amdgcn_wmma_f32_16x16x32_f16(false, aQ[1], false, b1,
                                                          (short)0, cacc, false, false);
            st[nt] = cacc;
        }

        float* Sw = Ssc[wv];
#pragma unroll
        for (int nt = 0; nt < 4; ++nt)
#pragma unroll
            for (int r = 0; r < 8; ++r)
                Sw[(8*g + r) * SSTR + nt*16 + m16] = st[nt][r];
        __builtin_amdgcn_wave_barrier();

        const float* srow = &Sw[m16 * SSTR + 32*g];
        float v[32];
#pragma unroll
        for (int q4 = 0; q4 < 8; ++q4) {
            float4 x = *reinterpret_cast<const float4*>(srow + 4*q4);
            v[4*q4+0]=x.x; v[4*q4+1]=x.y; v[4*q4+2]=x.z; v[4*q4+3]=x.w;
        }

        if (key0 + (KVT - 1) > qrow0) {
            const int c0 = key0 + 32*g;
#pragma unroll
            for (int j = 0; j < 32; ++j)
                if (c0 + j > grow) v[j] = -__builtin_inff();
        }

        float tm[16];
#pragma unroll
        for (int j = 0; j < 16; ++j) tm[j] = fmaxf(v[j], v[j+16]);
#pragma unroll
        for (int j = 0; j < 8; ++j)  tm[j] = fmaxf(tm[j], tm[j+8]);
#pragma unroll
        for (int j = 0; j < 4; ++j)  tm[j] = fmaxf(tm[j], tm[j+4]);
        float mx = fmaxf(fmaxf(tm[0], tm[1]), fmaxf(tm[2], tm[3]));
        mx = fmaxf(mx, __shfl_xor(mx, 16, 32));

        float mnew = fmaxf(mrow, mx);
#pragma unroll
        for (int j = 0; j < 32; ++j) v[j] = exp2f(v[j] - mnew);

        float ts[16];
#pragma unroll
        for (int j = 0; j < 16; ++j) ts[j] = v[j] + v[j+16];
#pragma unroll
        for (int j = 0; j < 8; ++j)  ts[j] = ts[j] + ts[j+8];
#pragma unroll
        for (int j = 0; j < 4; ++j)  ts[j] = ts[j] + ts[j+4];
        float ps = (ts[0] + ts[1]) + (ts[2] + ts[3]);
        ps += __shfl_xor(ps, 16, 32);

        float cr = exp2f(mrow - mnew);
        lrow = lrow * cr + ps;
        mrow = mnew;

        _Float16* Pw = Pst[wv];
#pragma unroll
        for (int q8 = 0; q8 < 4; ++q8) {
            v8h ph;
#pragma unroll
            for (int j = 0; j < 8; ++j) ph[j] = (_Float16)v[8*q8 + j];
            *reinterpret_cast<v8h*>(&Pw[m16 * PSTR + 32*g + 8*q8]) = ph;
        }

        if (g == 0) Cst[wv][m16] = cr;
        __builtin_amdgcn_wave_barrier();
        float4 ca = *reinterpret_cast<const float4*>(&Cst[wv][8*g]);
        float4 cb = *reinterpret_cast<const float4*>(&Cst[wv][8*g + 4]);
        float cf[8] = { ca.x,ca.y,ca.z,ca.w, cb.x,cb.y,cb.z,cb.w };
#pragma unroll
        for (int t = 0; t < 4; ++t)
#pragma unroll
            for (int r = 0; r < 8; ++r)
                oacc[t][r] *= cf[r];

        v16h aP[2];
#pragma unroll
        for (int c = 0; c < 2; ++c)
            aP[c] = cat16(*reinterpret_cast<const v8h*>(&Pw[m16 * PSTR + 32*c + 16*g]),
                          *reinterpret_cast<const v8h*>(&Pw[m16 * PSTR + 32*c + 16*g + 8]));

#pragma unroll
        for (int nt = 0; nt < 4; ++nt) {
            const _Float16* vrow = &Vt[cur][(nt*16 + m16) * VSTR];
            v16h b0 = cat16(*reinterpret_cast<const v8h*>(vrow + 16*g),
                            *reinterpret_cast<const v8h*>(vrow + 16*g + 8));
            v16h b1 = cat16(*reinterpret_cast<const v8h*>(vrow + 32 + 16*g),
                            *reinterpret_cast<const v8h*>(vrow + 32 + 16*g + 8));
            oacc[nt] = __builtin_amdgcn_wmma_f32_16x16x32_f16(false, aP[0], false, b0,
                                                              (short)0, oacc[nt], false, false);
            oacc[nt] = __builtin_amdgcn_wmma_f32_16x16x32_f16(false, aP[1], false, b1,
                                                              (short)0, oacc[nt], false, false);
        }

        if (have_next) stage_store(Kt[nxt], Vt[nxt], u, c4, kr, vr);
        __syncthreads();
    }

    if (g == 0) Cst[wv][m16] = 1.0f / lrow;
    __builtin_amdgcn_wave_barrier();
    float4 ia = *reinterpret_cast<const float4*>(&Cst[wv][8*g]);
    float4 ib = *reinterpret_cast<const float4*>(&Cst[wv][8*g + 4]);
    float inv[8] = { ia.x,ia.y,ia.z,ia.w, ib.x,ib.y,ib.z,ib.w };
#pragma unroll
    for (int nt = 0; nt < 4; ++nt)
#pragma unroll
        for (int r = 0; r < 8; ++r)
            Oh[(size_t)(qrow0 + 8*g + r) * DVV + nt*16 + m16] = oacc[nt][r] * inv[r];
}

extern "C" void kernel_launch(void* const* d_in, const int* in_sizes, int n_in,
                              void* d_out, int out_size, void* d_ws, size_t ws_size,
                              hipStream_t stream) {
    const float* Q = (const float*)d_in[0];
    const float* K = (const float*)d_in[1];
    const float* V = (const float*)d_in[2];
    float* Oout = (float*)d_out;
    (void)in_sizes; (void)n_in; (void)out_size;

    const size_t elems  = (size_t)NBH * S_LEN * DKK;         // 4.19M per tensor
    const size_t needed = 2 * elems * sizeof(_Float16);      // K16 + V16T = 16.8 MB

    dim3 grid(NBH * (S_LEN / QTILE));   // 512 workgroups
    dim3 block(256);                     // 8 waves, one 16-row Q tile each

    if (ws_size >= needed) {
        _Float16* K16  = (_Float16*)d_ws;
        _Float16* V16T = K16 + elems;
        cvt_kv_f16<<<dim3((unsigned)(elems / 4 / 256)), 256, 0, stream>>>(K, V, K16, V16T);
        fa_causal_f16wmma_async<<<grid, block, 0, stream>>>(Q, K16, V16T, Oout);
    } else {
        fa_causal_f16wmma<<<grid, block, 0, stream>>>(Q, K, V, Oout);
    }
}
